// LSTMModel_17197049053470
// MI455X (gfx1250) — matrix operand order repaired and checked by reading, not compiled, not run on hardware
//
#include <hip/hip_runtime.h>
#include <hip/hip_bf16.h>

// ---- CDNA5 (gfx1250) LSTM LM ----
// bf16 WMMA for all three GEMMs (fp32 accumulate).
// Vocab GEMM: LDS-staged B via global_load_async_to_lds_b128, ping-pong buffers.
// All pipelined loops in two-phase unrolled form (no rotation movs / zero-inits).
// Recurrence: 512 graph-captured launches, W_hh resident in L2.

typedef __attribute__((ext_vector_type(16))) __bf16 v16bf;
typedef __attribute__((ext_vector_type(8)))  __bf16 v8bf;
typedef __attribute__((ext_vector_type(8)))  float  v8f;

#define VOCAB 32000
#define EMB   1024
#define HID   1024
#define BATCH 8
#define SEQ   512
#define ROWS  (BATCH * SEQ)   // 4096
#define G4    (4 * HID)       // 4096

// gemm_out block tiling
#define BN    256             // block N tile
#define LPAD  8               // LDS row pad (bf16 elems) -> 528B row stride, no bank conflicts
#define BROW  (BN + LPAD)

static __device__ __forceinline__ v8f wmma_bf16(v16bf a, v16bf b, v8f c) {
  return __builtin_amdgcn_wmma_f32_16x16x32_bf16(false, a, false, b, (short)0, c, false, false);
}

// A fragment (16x32 bf16): lane -> row M = lane&15; half-wave selects K-base 0 or 8;
// elems [0..7] = K kb+0..7, [8..15] = K kb+16..23 (ISA 16-bit A layout).
static __device__ __forceinline__ v16bf load_a_bf16(const __bf16* __restrict__ A, int lda,
                                                    int row_base, int kc) {
  int lane = threadIdx.x & 31;
  int m  = lane & 15;
  int kb = (lane >> 4) * 8;
  const __bf16* p = A + (size_t)(row_base + m) * lda + kc + kb;
  v16bf a;
  ((v8bf*)&a)[0] = *(const v8bf*)(p);
  ((v8bf*)&a)[1] = *(const v8bf*)(p + 16);
  return a;
}

// A layout as above, gathering the fp32 embedding row of token x[row], cvt to bf16.
static __device__ __forceinline__ v16bf load_a_emb(const float* __restrict__ emb,
                                                   const int* __restrict__ x,
                                                   int row_base, int kc) {
  int lane = threadIdx.x & 31;
  int m  = lane & 15;
  int kb = (lane >> 4) * 8;
  int token = x[row_base + m];
  const float* p = emb + (size_t)token * EMB + kc + kb;
  v8f lo = *(const v8f*)p;
  v8f hi = *(const v8f*)(p + 16);
  v16bf a;
#pragma unroll
  for (int i = 0; i < 8; ++i) { a[i] = (__bf16)lo[i]; a[i + 8] = (__bf16)hi[i]; }
  return a;
}

// B fragment (32x16 bf16): lane -> col N = lane&15; elems [0..7] = K kb+0..7, [8..15] = K kb+16..23.
static __device__ __forceinline__ v16bf load_b(const __bf16* __restrict__ Bt, int ldn,
                                               int kc, int n0) {
  int lane = threadIdx.x & 31;
  int kb = (lane >> 4) * 8;
  const __bf16* p = Bt + (size_t)(kc + kb) * ldn + n0 + (lane & 15);
  v16bf b;
#pragma unroll
  for (int i = 0; i < 8; ++i) { b[i] = p[(size_t)i * ldn]; b[i + 8] = p[(size_t)(16 + i) * ldn]; }
  return b;
}

// B fragment from an LDS-staged slab (bp = row 0 of column lane&15; K in A's table order).
static __device__ __forceinline__ v16bf load_b_lds(const __bf16* bp) {
  int kb = ((threadIdx.x & 31) >> 4) * 8;
  v16bf b;
#pragma unroll
  for (int i = 0; i < 8; ++i) { b[i] = bp[(kb + i) * BROW]; b[i + 8] = bp[(kb + 16 + i) * BROW]; }
  return b;
}

// Stage a 32 x BN bf16 slab of B into LDS with async copies (256 threads, 4 x b128 each).
static __device__ __forceinline__ void stage_b_async(const __bf16* __restrict__ src,
                                                     unsigned lds_base, int ldn,
                                                     int kc, int n0) {
  const int tid = threadIdx.x;  // 0..255
#pragma unroll
  for (int i = 0; i < 4; ++i) {
    int j    = i * 256 + tid;       // 0..1023 b128 words
    int row  = j >> 5;              // 0..31 (32 b128 per 256-elem row)
    int cole = (j & 31) * 8;        // element column (8 bf16 per b128)
    unsigned long long ga =
        (unsigned long long)(uintptr_t)(src + (size_t)(kc + row) * ldn + n0 + cole);
    unsigned la = lds_base + (unsigned)(row * (BROW * 2) + cole * 2);
    asm volatile("global_load_async_to_lds_b128 %0, %1, off"
                 :: "v"(la), "v"(ga) : "memory");
  }
}

static __device__ __forceinline__ void wait_async_all() {
  asm volatile("s_wait_asynccnt 0x0" ::: "memory");
}

// ---- one-time fp32 [R][C] -> bf16 [C][R] transpose/convert (LDS tiled) ----
__global__ void transpose_to_bf16(const float* __restrict__ in, __bf16* __restrict__ out,
                                  int R, int C) {
  __shared__ float tile[32][33];
  int c0 = blockIdx.x * 32;
  int r0 = blockIdx.y * 32;
  int tx = threadIdx.x, ty = threadIdx.y;  // 32 x 8
#pragma unroll
  for (int k = 0; k < 32; k += 8)
    tile[ty + k][tx] = in[(size_t)(r0 + ty + k) * C + c0 + tx];
  __syncthreads();
#pragma unroll
  for (int k = 0; k < 32; k += 8)
    out[(size_t)(c0 + ty + k) * R + r0 + tx] = (__bf16)tile[tx][ty + k];
}

__global__ void init_state(__bf16* __restrict__ h0, __bf16* __restrict__ h1,
                           float* __restrict__ c) {
  int i = blockIdx.x * blockDim.x + threadIdx.x;
  if (i < 16 * HID) { h0[i] = (__bf16)0.f; h1[i] = (__bf16)0.f; }
  if (i < BATCH * HID) c[i] = 0.f;
}

// ---- x_gates = emb[x] @ W_ih^T + (b_ih + b_hh), [4096 x 4096], K = 1024 ----
// Two-phase pipelined: phase-1 fragments load while phase-0 WMMAs run, and vice versa.
__global__ void __launch_bounds__(256)
gemm_xgates(const float* __restrict__ emb, const int* __restrict__ x,
            const __bf16* __restrict__ WihT,
            const float* __restrict__ b_ih, const float* __restrict__ b_hh,
            float* __restrict__ xg) {
  int w = blockIdx.x * (blockDim.x >> 5) + (threadIdx.x >> 5);
  const int NB = G4 / 64;
  int mt = w / NB, nb = w % NB;
  int row_base = mt * 16;
  int n0 = nb * 64;
  v8f acc[4] = {};
  v16bf a0 = load_a_emb(emb, x, row_base, 0);
  v16bf b0[4], a1, b1[4];
#pragma unroll
  for (int t = 0; t < 4; ++t) b0[t] = load_b(WihT, G4, 0, n0 + t * 16);

  for (int kc = 0; kc < EMB; kc += 64) {
    a1 = load_a_emb(emb, x, row_base, kc + 32);
#pragma unroll
    for (int t = 0; t < 4; ++t) b1[t] = load_b(WihT, G4, kc + 32, n0 + t * 16);
#pragma unroll
    for (int t = 0; t < 4; ++t) acc[t] = wmma_bf16(a0, b0[t], acc[t]);
    if (kc + 64 < EMB) {
      __builtin_prefetch(WihT + (size_t)(kc + 96 + (threadIdx.x & 31)) * G4 + n0, 0, 1);
      a0 = load_a_emb(emb, x, row_base, kc + 64);
#pragma unroll
      for (int t = 0; t < 4; ++t) b0[t] = load_b(WihT, G4, kc + 64, n0 + t * 16);
    }
#pragma unroll
    for (int t = 0; t < 4; ++t) acc[t] = wmma_bf16(a1, b1[t], acc[t]);
  }
  int lane = threadIdx.x & 31;
  int m = 8 * (lane >> 4), n = lane & 15;
#pragma unroll
  for (int t = 0; t < 4; ++t) {
    float bias = b_ih[n0 + t * 16 + n] + b_hh[n0 + t * 16 + n];
#pragma unroll
    for (int v = 0; v < 8; ++v)
      xg[(size_t)(row_base + m + v) * G4 + n0 + t * 16 + n] = acc[t][v] + bias;
  }
}

// ---- one recurrence step: gates = xg[t] + h @ W_hh^T; activations; c,h update ----
// Two-phase pipeline + split even/odd accumulators (halves the WMMA RAW chain,
// this kernel is the latency-critical sequential part).
__global__ void __launch_bounds__(128)
lstm_step(const __bf16* __restrict__ h_in, __bf16* __restrict__ h_out,
          float* __restrict__ c, const __bf16* __restrict__ WhhT,
          const float* __restrict__ xg, __bf16* __restrict__ lout, int t) {
  __shared__ float gl[4][16][16];
  int wave = threadIdx.x >> 5;          // gate 0..3 (i,f,g,o)
  int lane = threadIdx.x & 31;
  int u0 = blockIdx.x * 16;             // hidden-unit base
  int n0 = wave * HID + u0;             // column in [0, 4H)
  v8f acc0 = {}, acc1 = {};
  v16bf a0 = load_a_bf16(h_in, HID, 0, 0);    // batch padded 8->16, rows 8..15 stay zero
  v16bf b0 = load_b(WhhT, G4, 0, n0);
  v16bf a1, b1;
  for (int kc = 0; kc < HID; kc += 64) {
    a1 = load_a_bf16(h_in, HID, 0, kc + 32);
    b1 = load_b(WhhT, G4, kc + 32, n0);
    acc0 = wmma_bf16(a0, b0, acc0);
    if (kc + 64 < HID) {
      a0 = load_a_bf16(h_in, HID, 0, kc + 64);
      b0 = load_b(WhhT, G4, kc + 64, n0);
    }
    acc1 = wmma_bf16(a1, b1, acc1);
  }
  int m = 8 * (lane >> 4), n = lane & 15;
#pragma unroll
  for (int v = 0; v < 8; ++v)
    gl[wave][m + v][n] = acc0[v] + acc1[v];
  __syncthreads();

  int b_ = threadIdx.x >> 4;            // 0..7 real batches
  int j  = threadIdx.x & 15;
  size_t r = (size_t)b_ * SEQ + t;
  float xi = xg[r * G4 + 0 * HID + u0 + j] + gl[0][b_][j];
  float xf = xg[r * G4 + 1 * HID + u0 + j] + gl[1][b_][j];
  float xn = xg[r * G4 + 2 * HID + u0 + j] + gl[2][b_][j];
  float xo = xg[r * G4 + 3 * HID + u0 + j] + gl[3][b_][j];
  float i_ = 1.f / (1.f + __expf(-xi));
  float f_ = 1.f / (1.f + __expf(-xf));
  float g_ = tanhf(xn);
  float o_ = 1.f / (1.f + __expf(-xo));
  float cc = f_ * c[b_ * HID + u0 + j] + i_ * g_;
  c[b_ * HID + u0 + j] = cc;
  float hh = o_ * tanhf(cc);
  h_out[b_ * HID + u0 + j] = (__bf16)hh;
  lout[r * HID + u0 + j]   = (__bf16)hh;
}

// ---- logits = lstm_out @ fc_w^T + fc_b, [4096 x 32000], K = 1024 (dominant GEMM) ----
// Block tile 32x256, 8 waves (2 M-tiles x 4 N-groups). B async-staged into LDS,
// ping-pong: stage buf1 / compute buf0 / sync / stage buf0 / compute buf1 / sync.
__global__ void __launch_bounds__(256)
gemm_out(const __bf16* __restrict__ Abf, const __bf16* __restrict__ fcwT,
         const float* __restrict__ fc_b, float* __restrict__ out) {
  __shared__ __bf16 Bs[2][32][BROW];    // 2 x 16.5 KB
  const int bid = blockIdx.x;
  const int bm  = bid / (VOCAB / BN);
  const int bn  = bid % (VOCAB / BN);
  const int w    = threadIdx.x >> 5;
  const int lane = threadIdx.x & 31;
  const int mt = w >> 2, ng = w & 3;
  const int row_base = bm * 32 + mt * 16;
  const int n0blk    = bn * BN;
  const int nw       = ng * 64;         // wave's N offset inside block tile
  const unsigned lbase = (unsigned)(uintptr_t)&Bs[0][0][0];
  const unsigned bufbytes = (unsigned)(32 * BROW * 2);

  // prologue: stage slab 0 into buf0, preload A fragment 0
  stage_b_async(fcwT, lbase, VOCAB, 0, n0blk);
  v16bf a0 = load_a_bf16(Abf, HID, row_base, 0);
  v16bf a1;
  wait_async_all();
  __syncthreads();

  v8f acc[4] = {};
  for (int kc = 0; kc < HID; kc += 64) {
    // stage (kc+32) into buf1 while computing buf0
    stage_b_async(fcwT, lbase + bufbytes, VOCAB, kc + 32, n0blk);
    a1 = load_a_bf16(Abf, HID, row_base, kc + 32);
#pragma unroll
    for (int t = 0; t < 4; ++t)
      acc[t] = wmma_bf16(a0, load_b_lds(&Bs[0][0][nw + t * 16 + (lane & 15)]), acc[t]);
    wait_async_all();    // buf1 ready...
    __syncthreads();     // ...everywhere; all waves done reading buf0

    // stage (kc+64) into buf0 while computing buf1
    if (kc + 64 < HID) {
      stage_b_async(fcwT, lbase, VOCAB, kc + 64, n0blk);
      a0 = load_a_bf16(Abf, HID, row_base, kc + 64);
    }
#pragma unroll
    for (int t = 0; t < 4; ++t)
      acc[t] = wmma_bf16(a1, load_b_lds(&Bs[1][0][nw + t * 16 + (lane & 15)]), acc[t]);
    wait_async_all();
    __syncthreads();
  }

  int m = 8 * (lane >> 4), n = lane & 15;
#pragma unroll
  for (int t = 0; t < 4; ++t) {
    int n0 = n0blk + nw + t * 16;
    float bias = fc_b[n0 + n];
#pragma unroll
    for (int v = 0; v < 8; ++v)
      out[(size_t)(row_base + m + v) * VOCAB + n0 + n] = acc[t][v] + bias;
  }
}

extern "C" void kernel_launch(void* const* d_in, const int* in_sizes, int n_in,
                              void* d_out, int out_size, void* d_ws, size_t ws_size,
                              hipStream_t stream) {
  const int*   x    = (const int*)d_in[0];
  const float* emb  = (const float*)d_in[1];
  const float* W_ih = (const float*)d_in[2];
  const float* W_hh = (const float*)d_in[3];
  const float* b_ih = (const float*)d_in[4];
  const float* b_hh = (const float*)d_in[5];
  const float* fc_w = (const float*)d_in[6];
  const float* fc_b = (const float*)d_in[7];
  float* out = (float*)d_out;

  // workspace carve-up (~151 MB total)
  char*   ws   = (char*)d_ws;
  size_t  o    = 0;
  __bf16* WihT = (__bf16*)(ws + o);  o += (size_t)EMB * G4 * 2;      //  8.4 MB  [E][4H]
  __bf16* WhhT = (__bf16*)(ws + o);  o += (size_t)HID * G4 * 2;      //  8.4 MB  [H][4H]
  __bf16* fcwT = (__bf16*)(ws + o);  o += (size_t)HID * VOCAB * 2;   // 65.5 MB  [H][V]
  float*  xg   = (float*)(ws + o);   o += (size_t)ROWS * G4 * 4;     // 67.1 MB  [B*S][4H]
  __bf16* lout = (__bf16*)(ws + o);  o += (size_t)ROWS * HID * 2;    //  8.4 MB  [B*S][H]
  __bf16* h0   = (__bf16*)(ws + o);  o += (size_t)16 * HID * 2;      //  padded h (ping)
  __bf16* h1   = (__bf16*)(ws + o);  o += (size_t)16 * HID * 2;      //  padded h (pong)
  float*  c    = (float*)(ws + o);   o += (size_t)BATCH * HID * 4;   //  cell state

  dim3 tb(32, 8);
  transpose_to_bf16<<<dim3(EMB / 32, G4 / 32),    tb, 0, stream>>>(W_ih, WihT, G4,    EMB);
  transpose_to_bf16<<<dim3(HID / 32, G4 / 32),    tb, 0, stream>>>(W_hh, WhhT, G4,    HID);
  transpose_to_bf16<<<dim3(HID / 32, VOCAB / 32), tb, 0, stream>>>(fc_w, fcwT, VOCAB, HID);
  init_state<<<(16 * HID + 255) / 256, 256, 0, stream>>>(h0, h1, c);

  { // input-side gate projections, hoisted out of the scan
    int waves = (ROWS / 16) * (G4 / 64);  // 16384
    gemm_xgates<<<waves / 8, 256, 0, stream>>>(emb, x, WihT, b_ih, b_hh, xg);
  }

  for (int t = 0; t < SEQ; ++t) {  // sequential recurrence; launch boundary = global sync
    const __bf16* hi = (t & 1) ? h1 : h0;
    __bf16*       ho = (t & 1) ? h0 : h1;
    lstm_step<<<HID / 16, 128, 0, stream>>>(hi, ho, c, WhhT, xg, lout, t);
  }

  { // vocab projection (dominant GEMM)
    int blocks = (ROWS / 32) * (VOCAB / BN);  // 128 * 125 = 16000
    gemm_out<<<blocks, 256, 0, stream>>>(lout, fcwT, fc_b, out);
  }
}
